// DeepGPCM_45037027066306
// MI455X (gfx1250) — compile-verified, matrix operand order checked by reading, not run on hardware
//
#include <hip/hip_runtime.h>
#include <hip/hip_bf16.h>

typedef __attribute__((ext_vector_type(16))) _Float16 v16h;
typedef __attribute__((ext_vector_type(8)))  _Float16 v8h;
typedef __attribute__((ext_vector_type(8)))  float    v8f;

#define NB   128          // batch
#define NS   2048         // seq
#define NROWS (NB*NS)     // 262144
#define MQ   50           // memory slots
#define DK   64
#define DV   128
#define DS   50

// ---------------- workspace layout (bytes) ----------------
static constexpr size_t OFF_QE    = 0;                                  // N*64 f16
static constexpr size_t OFF_VIN   = OFF_QE    + (size_t)NROWS*64*2;     // N*96 f16
static constexpr size_t OFF_ATTN  = OFF_VIN   + (size_t)NROWS*96*2;     // N*50 f32
static constexpr size_t OFF_ERASE = OFF_ATTN  + (size_t)NROWS*50*4;     // N*128 f16
static constexpr size_t OFF_ADD   = OFF_ERASE + (size_t)NROWS*128*2;    // N*128 f16
static constexpr size_t OFF_READS = OFF_ADD   + (size_t)NROWS*128*2;    // N*128 f16
static constexpr size_t OFF_FKM   = OFF_READS + (size_t)NROWS*128*2;    // 2*4*512 f16
static constexpr size_t OFF_FVP   = OFF_FKM   + 2*4*512*2;              // 3*8*512 f16
static constexpr size_t OFF_FEW   = OFF_FVP   + 3*8*512*2;              // 4*8*512 f16
static constexpr size_t OFF_FAW   = OFF_FEW   + 4*8*512*2;              // 4*8*512 f16
static constexpr size_t OFF_FSW   = OFF_FAW   + 4*8*512*2;              // 6*4*512 f16

// ---------------- helpers ----------------
// A-matrix 16x32 f16 layout: lane l (row = l&15), elems 0..7 -> K = kh..kh+7,
// elems 8..15 -> K = 16+kh..16+kh+7, where kh = (l>>4)*8.  Two 16B loads.
__device__ inline v16h load_a16(const _Float16* p) {
    v8h lo = *(const v8h*)p;
    v8h hi = *(const v8h*)(p + 16);
    v16h r;
#pragma unroll
    for (int i = 0; i < 8; ++i) { r[i] = lo[i]; r[i + 8] = hi[i]; }
    return r;
}

// B fragments pre-packed by prep_kernel: ((kt*NT + nt)*32 + lane)*16 contiguous halves.
__device__ inline v16h load_b_frag(const _Float16* base, int kt, int nt, int NT, int lane) {
    return *(const v16h*)(base + ((size_t)((kt * NT + nt) * 32 + lane) << 4));
}

__device__ inline v8f wmma_step(v16h a, v16h b, v8f c) {
    return __builtin_amdgcn_wmma_f32_16x16x32_f16(false, a, false, b, (short)0, c, false, false);
}

// Lightweight workgroup barrier: order LDS only; leave vmem loads in flight.
__device__ inline void wg_barrier() {
    asm volatile("s_wait_dscnt 0x0" ::: "memory");
    __builtin_amdgcn_s_barrier();
}

__device__ inline void cvt_store8(_Float16* dst, const float* src) {
    const float4* s4 = (const float4*)src;
    float4 a = s4[0], b = s4[1];
    union { _Float16 h[8]; uint4 u; } t;
    t.h[0] = (_Float16)a.x; t.h[1] = (_Float16)a.y; t.h[2] = (_Float16)a.z; t.h[3] = (_Float16)a.w;
    t.h[4] = (_Float16)b.x; t.h[5] = (_Float16)b.y; t.h[6] = (_Float16)b.z; t.h[7] = (_Float16)b.w;
    *(uint4*)dst = t.u;
}

// ---------------- weight repack into B-fragment order ----------------
__device__ void fill_frag(_Float16* dst, const float* src, int Ksrc, int Nsrc,
                          int KT, int NT, int transpose, int tid, int nth) {
    int total = KT * NT * 512;
    for (int idx = tid; idx < total; idx += nth) {
        int e = idx & 15, lane = (idx >> 4) & 31, t = idx >> 9;
        int nt = t % NT, kt = t / NT;
        int r16 = lane & 15, h = lane >> 4;
        int k = kt * 32 + ((e < 8) ? (h * 8 + e) : (16 + h * 8 + (e - 8)));
        int n = nt * 16 + r16;
        float v = 0.f;
        if (k < Ksrc && n < Nsrc)
            v = transpose ? src[(size_t)n * Ksrc + k] : src[(size_t)k * Nsrc + n];
        dst[idx] = (_Float16)v;
    }
}

__global__ void prep_kernel(const float* km, const float* vp, const float* ew,
                            const float* aw, const float* sw,
                            _Float16* fk, _Float16* fv, _Float16* fe,
                            _Float16* fa, _Float16* fs) {
    int tid = blockIdx.x * blockDim.x + threadIdx.x;
    int nth = gridDim.x * blockDim.x;
    fill_frag(fk, km, 64, 50, 2, 4, 1, tid, nth);     // B = key_mem^T
    fill_frag(fv, vp, 69, 128, 3, 8, 0, tid, nth);    // value_proj_w
    fill_frag(fe, ew, 128, 128, 4, 8, 0, tid, nth);   // erase_w
    fill_frag(fa, aw, 128, 128, 4, 8, 0, tid, nth);   // add_w
    fill_frag(fs, sw, 192, 50, 6, 4, 0, tid, nth);    // summary_w
}

// ---------------- gather + response features (vectorized) ----------------
__global__ __launch_bounds__(256) void gather_embed(const int* questions, const int* responses,
                                                    const float* qw, const float* iw,
                                                    _Float16* qe, _Float16* vin) {
    int row = blockIdx.x * blockDim.x + threadIdx.x;
    if (row >= NROWS) return;
    int q = questions[row];
    float r = (float)responses[row];
    const float* qs = qw + (size_t)q * 64;
    const float* is = iw + (size_t)q * 64;
    _Float16* qd = qe + (size_t)row * 64;
    _Float16* vd = vin + (size_t)row * 96;
#pragma unroll
    for (int c = 0; c < 8; ++c) {
        cvt_store8(qd + c * 8, qs + c * 8);
        cvt_store8(vd + c * 8, is + c * 8);
    }
    union { _Float16 h[8]; uint4 u; } tbuf;
#pragma unroll
    for (int i = 0; i < 8; ++i) {
        float f = (i < 5) ? fmaxf(1.f - fabsf((float)i - r) * 0.25f, 0.f) : 0.f;
        tbuf.h[i] = (_Float16)f;
    }
    *(uint4*)(vd + 64) = tbuf.u;
    uint4 z = make_uint4(0, 0, 0, 0);
    *(uint4*)(vd + 72) = z;
    *(uint4*)(vd + 80) = z;
    *(uint4*)(vd + 88) = z;
}

// ---------------- attention logits (WMMA) + softmax ----------------
__global__ __launch_bounds__(256) void attn_kernel(const _Float16* qe, const _Float16* fkm,
                                                   float* attn) {
    __shared__ float lds[128][65];
    int wave = threadIdx.x >> 5, lane = threadIdx.x & 31;
    int r16 = lane & 15, kh = (lane >> 4) * 8;
    size_t rowBase = (size_t)blockIdx.x * 128 + wave * 16;

    v16h a[2];
#pragma unroll
    for (int kt = 0; kt < 2; ++kt)
        a[kt] = load_a16(qe + (rowBase + r16) * 64 + kt * 32 + kh);

    v8f c[4];
#pragma unroll
    for (int nt = 0; nt < 4; ++nt)
#pragma unroll
        for (int i = 0; i < 8; ++i) c[nt][i] = 0.f;

#pragma unroll
    for (int kt = 0; kt < 2; ++kt) {
        v16h bf[4];
#pragma unroll
        for (int nt = 0; nt < 4; ++nt) bf[nt] = load_b_frag(fkm, kt, nt, 4, lane);
#pragma unroll
        for (int nt = 0; nt < 4; ++nt) c[nt] = wmma_step(a[kt], bf[nt], c[nt]);
    }
#pragma unroll
    for (int nt = 0; nt < 4; ++nt)
#pragma unroll
        for (int v = 0; v < 8; ++v) {
            int rr = wave * 16 + ((lane < 16) ? v : v + 8);
            lds[rr][nt * 16 + r16] = c[nt][v];
        }
    __syncthreads();

    int tid = threadIdx.x;
    if (tid < 128) {
        float vals[50], mx = -1e30f;
#pragma unroll 5
        for (int i = 0; i < 50; ++i) { vals[i] = lds[tid][i]; mx = fmaxf(mx, vals[i]); }
        float s = 0.f;
#pragma unroll 5
        for (int i = 0; i < 50; ++i) { vals[i] = __expf(vals[i] - mx); s += vals[i]; }
        float inv = 1.f / s;
        float* out = attn + ((size_t)blockIdx.x * 128 + tid) * 50;
#pragma unroll 5
        for (int i = 0; i < 50; ++i) out[i] = vals[i] * inv;
    }
}

// ---------------- value_embed -> erase/add (fused, WMMA, per-wave slabs) ----------------
__global__ __launch_bounds__(256) void vea_kernel(const _Float16* vin, const _Float16* fvp,
                                                  const _Float16* few, const _Float16* faw,
                                                  const float* vpb, const float* eb, const float* ab,
                                                  _Float16* eraseg, _Float16* addg) {
    __shared__ __align__(16) _Float16 ve[8][16][128];
    __shared__ __align__(16) _Float16 ot[8][16][128];
    int wave = threadIdx.x >> 5, lane = threadIdx.x & 31;
    int r16 = lane & 15, kh = (lane >> 4) * 8;
    size_t rowBase = (size_t)blockIdx.x * 128 + wave * 16;

    // GEMM1: value_embed = vin(16x96) @ Wv(96x128) + b
    v16h a1[3];
#pragma unroll
    for (int kt = 0; kt < 3; ++kt)
        a1[kt] = load_a16(vin + (rowBase + r16) * 96 + kt * 32 + kh);

    v8f c[8];
#pragma unroll
    for (int nt = 0; nt < 8; ++nt) {
        float bias = vpb[nt * 16 + r16];
#pragma unroll
        for (int i = 0; i < 8; ++i) c[nt][i] = bias;
    }
#pragma unroll
    for (int kt = 0; kt < 3; ++kt) {
        v16h bf[8];
#pragma unroll
        for (int nt = 0; nt < 8; ++nt) bf[nt] = load_b_frag(fvp, kt, nt, 8, lane);
#pragma unroll
        for (int nt = 0; nt < 8; ++nt) c[nt] = wmma_step(a1[kt], bf[nt], c[nt]);
    }
#pragma unroll
    for (int nt = 0; nt < 8; ++nt)
#pragma unroll
        for (int v = 0; v < 8; ++v) {
            int rr = (lane < 16) ? v : v + 8;
            ve[wave][rr][nt * 16 + r16] = (_Float16)c[nt][v];
        }

    // per-wave slab: same-wave DS ops are in-order; no block barrier needed
    v16h a2[4];
#pragma unroll
    for (int kt = 0; kt < 4; ++kt)
        a2[kt] = load_a16(&ve[wave][r16][kt * 32 + kh]);

    // GEMM2: erase = sigmoid(ve @ We + be)
#pragma unroll
    for (int nt = 0; nt < 8; ++nt) {
        float bias = eb[nt * 16 + r16];
#pragma unroll
        for (int i = 0; i < 8; ++i) c[nt][i] = bias;
    }
#pragma unroll
    for (int kt = 0; kt < 4; ++kt) {
        v16h bf[8];
#pragma unroll
        for (int nt = 0; nt < 8; ++nt) bf[nt] = load_b_frag(few, kt, nt, 8, lane);
#pragma unroll
        for (int nt = 0; nt < 8; ++nt) c[nt] = wmma_step(a2[kt], bf[nt], c[nt]);
    }
#pragma unroll
    for (int nt = 0; nt < 8; ++nt)
#pragma unroll
        for (int v = 0; v < 8; ++v) {
            int rr = (lane < 16) ? v : v + 8;
            ot[wave][rr][nt * 16 + r16] = (_Float16)(1.f / (1.f + __expf(-c[nt][v])));
        }
    {   // coalesced copy of this wave's 16x128 f16 tile
        const uint4* s = (const uint4*)&ot[wave][0][0];
        uint4* d = (uint4*)(eraseg + rowBase * 128);
        for (int i = lane; i < 256; i += 32) d[i] = s[i];
    }

    // GEMM3: add = tanh(ve @ Wa + ba)
#pragma unroll
    for (int nt = 0; nt < 8; ++nt) {
        float bias = ab[nt * 16 + r16];
#pragma unroll
        for (int i = 0; i < 8; ++i) c[nt][i] = bias;
    }
#pragma unroll
    for (int kt = 0; kt < 4; ++kt) {
        v16h bf[8];
#pragma unroll
        for (int nt = 0; nt < 8; ++nt) bf[nt] = load_b_frag(faw, kt, nt, 8, lane);
#pragma unroll
        for (int nt = 0; nt < 8; ++nt) c[nt] = wmma_step(a2[kt], bf[nt], c[nt]);
    }
#pragma unroll
    for (int nt = 0; nt < 8; ++nt)
#pragma unroll
        for (int v = 0; v < 8; ++v) {
            int rr = (lane < 16) ? v : v + 8;
            ot[wave][rr][nt * 16 + r16] = (_Float16)tanhf(c[nt][v]);
        }
    {
        const uint4* s = (const uint4*)&ot[wave][0][0];
        uint4* d = (uint4*)(addg + rowBase * 128);
        for (int i = lane; i < 256; i += 32) d[i] = s[i];
    }
}

// ---------------- sequential memory scan (LDS mem, register erase/add, pipelined) ----------------
__global__ __launch_bounds__(256) void scan_kernel(const float* init_mem, const float* attn,
                                                   const _Float16* eraseg, const _Float16* addg,
                                                   _Float16* reads) {
    __shared__ float mem[MQ][DV];
    __shared__ float at[2][MQ];
    int b = blockIdx.x, tid = threadIdx.x;
    int v = tid & 127;           // each thread owns column v of the memory
    int mh = tid >> 7;           // row-parity this thread updates
    size_t row0 = (size_t)b * NS;

    for (int i = tid; i < MQ * DV; i += 256) mem[i / DV][i % DV] = init_mem[i];

    // stage t = 0
    float er_c = (float)eraseg[row0 * 128 + v];
    float ad_c = (float)addg[row0 * 128 + v];
    if (tid < MQ) at[0][tid] = attn[row0 * 50 + tid];
    wg_barrier();

    for (int t = 0; t < NS; ++t) {
        int cur = t & 1, nxt = cur ^ 1;
        size_t row = row0 + t;

        // issue t+1 loads early; only waited on at the staging store below
        float er_n = 0.f, ad_n = 0.f, at_n = 0.f;
        if (t + 1 < NS) {
            size_t rn = row + 1;
            er_n = (float)eraseg[rn * 128 + v];
            ad_n = (float)addg[rn * 128 + v];
            if (tid < MQ) at_n = attn[rn * 50 + tid];
            if (t + 2 < NS) {
                __builtin_prefetch(eraseg + (rn + 1) * 128 + v);
                __builtin_prefetch(addg + (rn + 1) * 128 + v);
            }
        }

        if (tid < 128) {
            float rd = 0.f;
#pragma unroll 5
            for (int m = 0; m < MQ; ++m) rd += at[cur][m] * mem[m][tid];
            reads[row * 128 + tid] = (_Float16)rd;
        }
        wg_barrier();                       // reads of mem done before updates

        for (int m = mh; m < MQ; m += 2) {
            float w = at[cur][m];
            mem[m][v] = mem[m][v] * (1.f - w * er_c) + w * ad_c;
        }
        if (tid < MQ) at[nxt][tid] = at_n;  // different buffer than at[cur]
        er_c = er_n; ad_c = ad_n;
        wg_barrier();                       // updates + staging done
    }
}

// ---------------- summary GEMM + GPCM heads ----------------
__global__ __launch_bounds__(256) void head_kernel(const _Float16* reads, const _Float16* qe,
                                                   const _Float16* fsw, const float* sb,
                                                   const float* tw, const float* tb,
                                                   const float* aw, const float* abv,
                                                   const float* bw, const float* bb,
                                                   float* out) {
    __shared__ float summ[128][65];
    int wave = threadIdx.x >> 5, lane = threadIdx.x & 31;
    int r16 = lane & 15, kh = (lane >> 4) * 8;
    size_t rowBase = (size_t)blockIdx.x * 128 + wave * 16;

    v16h a[6];
#pragma unroll
    for (int kt = 0; kt < 4; ++kt)
        a[kt] = load_a16(reads + (rowBase + r16) * 128 + kt * 32 + kh);
#pragma unroll
    for (int kt = 4; kt < 6; ++kt)
        a[kt] = load_a16(qe + (rowBase + r16) * 64 + (kt - 4) * 32 + kh);

    v8f c[4];
#pragma unroll
    for (int nt = 0; nt < 4; ++nt) {
        int ci = nt * 16 + r16;
        float bias = (ci < 50) ? sb[ci] : 0.f;
#pragma unroll
        for (int i = 0; i < 8; ++i) c[nt][i] = bias;
    }
#pragma unroll
    for (int kt = 0; kt < 6; ++kt) {
        v16h bf[4];
#pragma unroll
        for (int nt = 0; nt < 4; ++nt) bf[nt] = load_b_frag(fsw, kt, nt, 4, lane);
#pragma unroll
        for (int nt = 0; nt < 4; ++nt) c[nt] = wmma_step(a[kt], bf[nt], c[nt]);
    }
#pragma unroll
    for (int nt = 0; nt < 4; ++nt)
#pragma unroll
        for (int v = 0; v < 8; ++v) {
            int rr = wave * 16 + ((lane < 16) ? v : v + 8);
            summ[rr][nt * 16 + r16] = tanhf(c[nt][v]);
        }
    __syncthreads();

    int tid = threadIdx.x;
    if (tid < 128) {
        size_t row = (size_t)blockIdx.x * 128 + tid;
        float th = tb[0];
#pragma unroll 5
        for (int i = 0; i < 50; ++i) th += summ[tid][i] * tw[i];
        th = tanhf(th);

        const _Float16* qp = qe + row * 64;
        float al = abv[0];
        float b0 = bb[0], b1 = bb[1], b2 = bb[2], b3 = bb[3];
#pragma unroll 4
        for (int i = 0; i < 64; ++i) {
            float q = (float)qp[i];
            al += q * aw[i];
            const float* bwi = bw + i * 4;
            b0 += q * bwi[0]; b1 += q * bwi[1]; b2 += q * bwi[2]; b3 += q * bwi[3];
        }
        al = (al > 20.f) ? al : __logf(1.f + __expf(al));   // softplus
        float inter = th * al;
        float lg[5];
        lg[0] = 0.f;
        lg[1] = inter - b0;
        lg[2] = lg[1] + inter - b1;
        lg[3] = lg[2] + inter - b2;
        lg[4] = lg[3] + inter - b3;
        float mx = lg[0];
#pragma unroll
        for (int j = 1; j < 5; ++j) mx = fmaxf(mx, lg[j]);
        float pr[5], s = 0.f;
#pragma unroll
        for (int j = 0; j < 5; ++j) { pr[j] = __expf(lg[j] - mx); s += pr[j]; }
        float inv = 1.f / s;

        const size_t N = (size_t)NROWS;
        out[row] = th;                                   // theta
        out[N + row] = al;                               // alpha
        float bevals[4] = {b0, b1, b2, b3};
#pragma unroll
        for (int j = 0; j < 4; ++j) out[2 * N + row * 4 + j] = bevals[j];    // beta
#pragma unroll
        for (int j = 0; j < 5; ++j) out[6 * N + row * 5 + j] = lg[j];        // logits
#pragma unroll
        for (int j = 0; j < 5; ++j) out[11 * N + row * 5 + j] = pr[j] * inv; // probs
    }
}

// ---------------- launcher ----------------
extern "C" void kernel_launch(void* const* d_in, const int* in_sizes, int n_in,
                              void* d_out, int out_size, void* d_ws, size_t ws_size,
                              hipStream_t stream) {
    const int*   questions    = (const int*)d_in[0];
    const int*   responses    = (const int*)d_in[1];
    const float* q_embed_w    = (const float*)d_in[2];
    const float* item_embed_w = (const float*)d_in[3];
    const float* value_proj_w = (const float*)d_in[4];
    const float* value_proj_b = (const float*)d_in[5];
    const float* key_mem      = (const float*)d_in[6];
    const float* init_mem     = (const float*)d_in[7];
    const float* erase_w      = (const float*)d_in[8];
    const float* erase_b      = (const float*)d_in[9];
    const float* add_w        = (const float*)d_in[10];
    const float* add_b        = (const float*)d_in[11];
    const float* summary_w    = (const float*)d_in[12];
    const float* summary_b    = (const float*)d_in[13];
    const float* theta_w      = (const float*)d_in[14];
    const float* theta_b      = (const float*)d_in[15];
    const float* alpha_w      = (const float*)d_in[16];
    const float* alpha_b      = (const float*)d_in[17];
    const float* beta_w       = (const float*)d_in[18];
    const float* beta_b       = (const float*)d_in[19];

    char* ws = (char*)d_ws;
    _Float16* qe_h   = (_Float16*)(ws + OFF_QE);
    _Float16* vin_h  = (_Float16*)(ws + OFF_VIN);
    float*    attn_f = (float*)(ws + OFF_ATTN);
    _Float16* er_h   = (_Float16*)(ws + OFF_ERASE);
    _Float16* ad_h   = (_Float16*)(ws + OFF_ADD);
    _Float16* rd_h   = (_Float16*)(ws + OFF_READS);
    _Float16* fkm    = (_Float16*)(ws + OFF_FKM);
    _Float16* fvp    = (_Float16*)(ws + OFF_FVP);
    _Float16* few    = (_Float16*)(ws + OFF_FEW);
    _Float16* faw    = (_Float16*)(ws + OFF_FAW);
    _Float16* fsw    = (_Float16*)(ws + OFF_FSW);

    prep_kernel<<<64, 256, 0, stream>>>(key_mem, value_proj_w, erase_w, add_w, summary_w,
                                        fkm, fvp, few, faw, fsw);
    gather_embed<<<NROWS / 256, 256, 0, stream>>>(questions, responses, q_embed_w,
                                                  item_embed_w, qe_h, vin_h);
    attn_kernel<<<NROWS / 128, 256, 0, stream>>>(qe_h, fkm, attn_f);
    vea_kernel<<<NROWS / 128, 256, 0, stream>>>(vin_h, fvp, few, faw,
                                                value_proj_b, erase_b, add_b, er_h, ad_h);
    scan_kernel<<<NB, 256, 0, stream>>>(init_mem, attn_f, er_h, ad_h, rd_h);
    head_kernel<<<NROWS / 128, 256, 0, stream>>>(rd_h, qe_h, fsw, summary_b,
                                                 theta_w, theta_b, alpha_w, alpha_b,
                                                 beta_w, beta_b, (float*)d_out);
}
